// HGT_36155034698322
// MI455X (gfx1250) — matrix operand order repaired and checked by reading, not compiled, not run on hardware
//
#include <hip/hip_runtime.h>
#include <hip/hip_bf16.h>
#include <math.h>

// ---------------- problem constants (from reference) ----------------
#define NQN     50000
#define NAN_    50000
#define HIDDEN  128
#define EEDGES  400000
#define HEADS   8
#define DHEAD   16
#define ATT_SCALE 0.25f   // 1/sqrt(16)

typedef __bf16 v16bf __attribute__((ext_vector_type(16)));
typedef __bf16 v8bf  __attribute__((ext_vector_type(8)));
typedef float  v8f   __attribute__((ext_vector_type(8)));

__device__ __forceinline__ float gelu_f(float x) {
    return 0.5f * x * (1.0f + erff(x * 0.70710678118654752440f));
}

__device__ __forceinline__ void atomicMaxFloat(float* addr, float val) {
    // monotone bit trick: positive floats order as int, negative as reversed uint
    if (val >= 0.0f) atomicMax((int*)addr, __float_as_int(val));
    else             atomicMin((unsigned int*)addr, __float_as_uint(val));
}

__device__ __forceinline__ void atomicAddF32(float* p, float v) {
    __hip_atomic_fetch_add(p, v, __ATOMIC_RELAXED, __HIP_MEMORY_SCOPE_AGENT);
}

// ---------------- small weight-prep kernels ----------------
__global__ void conv_bf16(const float* __restrict__ in, __bf16* __restrict__ outw, int n) {
    int t = blockIdx.x * blockDim.x + threadIdx.x;
    if (t < n) outw[t] = (__bf16)in[t];
}

// Fold relation matrix into weight: Weff[c][h*16+f] = sum_d W[c][h*16+d]*ar[h][d][f]
// row r==128 folds the bias: beff[h*16+f] = sum_d b[h*16+d]*ar[h][d][f]
__global__ void fuse_rel(const float* __restrict__ w, const float* __restrict__ b,
                         const float* __restrict__ ar, __bf16* __restrict__ weff,
                         float* __restrict__ beff) {
    int t = blockIdx.x * blockDim.x + threadIdx.x;
    if (t >= 129 * 128) return;
    int r = t >> 7;
    int n = t & 127;
    int h = n >> 4, f = n & 15;
    const float* in  = (r < 128) ? (w + r * 128 + h * 16) : (b + h * 16);
    const float* arp = ar + h * 256 + f;
    float acc = 0.f;
#pragma unroll
    for (int d = 0; d < 16; ++d) acc += in[d] * arp[d * 16];
    if (r < 128) weff[t] = (__bf16)acc;
    else         beff[n] = acc;
}

// ---------------- WMMA GEMM: C[M,N] = act(A[M,128]) @ Wbf16[128,N] + bias, + gated skip ----
// block = 256 threads (8 waves). A tile (16 x 128) staged once in LDS as bf16 (gelu applied
// during staging). Wave w owns column tile w*16. grid.x = M/16 (M % 16 == 0 guaranteed).
#define LDS_STRIDE 136   // 128 + 8 bf16 pad: row stride 272 B -> conflict-free ds_load_b128
__global__ __launch_bounds__(256)
void gemm_bias_act(const float* __restrict__ A,
                   const __bf16* __restrict__ W,
                   const float* __restrict__ bias,
                   float* __restrict__ C,
                   const float* __restrict__ skipOld,   // may be null (then no blend)
                   const float* __restrict__ gatePtr,   // scalar skip logit (sigmoid here)
                   int N, int preGelu, int postRelu) {
    __shared__ __bf16 As[16 * LDS_STRIDE];
    const int tid     = threadIdx.x;
    const int rowTile = blockIdx.x << 4;

    // ---- stage A tile: 2048 f32 -> gelu? -> bf16 -> LDS (each thread 8 elements) ----
    {
        const int idx = tid << 3;            // 0..2040
        const int r   = idx >> 7;            // 0..15
        const int c   = idx & 127;
        const float* ap = A + (size_t)(rowTile + r) * HIDDEN + c;
        const float4 f0 = *(const float4*)(ap);
        const float4 f1 = *(const float4*)(ap + 4);
        float v[8] = {f0.x, f0.y, f0.z, f0.w, f1.x, f1.y, f1.z, f1.w};
        v8bf st;
#pragma unroll
        for (int i = 0; i < 8; ++i) {
            float f = v[i];
            if (preGelu) f = gelu_f(f);
            st[i] = (__bf16)f;
        }
        *(v8bf*)(As + r * LDS_STRIDE + c) = st;
    }
    __syncthreads();

    const int lane    = tid & 31;
    const int wave    = tid >> 5;
    const int colBase = wave * 16;
    if (colBase >= N) return;                // wave-uniform; EXEC all-ones at every WMMA
    const int m    = lane & 15;
    const int half = lane >> 4;

    v8f acc = {0.f, 0.f, 0.f, 0.f, 0.f, 0.f, 0.f, 0.f};
#pragma unroll
    for (int kt = 0; kt < 4; ++kt) {
        const int kbase = kt * 32 + half * 8;
        // A fragment from LDS: lane holds K = kbase..kbase+7 and kbase+16..kbase+23
        const v8bf lo = *(const v8bf*)(As + m * LDS_STRIDE + kbase);
        const v8bf hi = *(const v8bf*)(As + m * LDS_STRIDE + kbase + 16);
        const v16bf afrag = __builtin_shufflevector(lo, hi,
                               0, 1, 2, 3, 4, 5, 6, 7, 8, 9, 10, 11, 12, 13, 14, 15);
        // B fragment: lane holds column N = colBase+(lane&15), K = kbase..kbase+7 and kbase+16..kbase+23
        v16bf bfrag;
#pragma unroll
        for (int s = 0; s < 16; ++s)
            bfrag[s] = W[(size_t)(kbase + (s < 8 ? s : s + 8)) * N + colBase + m];
        acc = __builtin_amdgcn_wmma_f32_16x16x32_bf16(false, afrag, false, bfrag,
                                                      (short)0, acc, false, false);
    }

    const int n = colBase + m;
    const float bval = bias ? bias[n] : 0.f;
    float g = 0.f;
    if (gatePtr) g = 1.f / (1.f + __expf(-gatePtr[0]));
#pragma unroll
    for (int i = 0; i < 8; ++i) {
        int r   = rowTile + half * 8 + i;    // C layout: VGPR i -> M = i + 8*half, N = lane&15
        float v = acc[i] + bval;
        if (postRelu) v = fmaxf(v, 0.f);
        if (skipOld)  v = g * v + (1.f - g) * skipOld[(size_t)r * N + n];
        C[(size_t)r * N + n] = v;
    }
}

// ---------------- edge-phase kernels ----------------
__global__ void init_msg(float* __restrict__ outm, float* __restrict__ maxb,
                         float* __restrict__ denb, int nNodes) {
    int t = blockIdx.x * blockDim.x + threadIdx.x;
    if (t < nNodes * HIDDEN) outm[t] = 0.f;
    if (t < nNodes * HEADS) { maxb[t] = -INFINITY; denb[t] = 0.f; }
}

// pass 1: alpha[e,h] = <q[dst,h,:], kt[src,h,:]> * pr[h] * scale ; segment max via atomics
__global__ __launch_bounds__(256)
void edge_alpha_max(const float* __restrict__ q, const float* __restrict__ kt,
                    const int* __restrict__ src, const int* __restrict__ dst,
                    const float* __restrict__ prel,
                    float* __restrict__ alpha, float* __restrict__ maxb, int nE) {
    int t = blockIdx.x * blockDim.x + threadIdx.x;
    if (t >= nE * HEADS) return;
    int e = t >> 3, h = t & 7;
    int s = src[e], d = dst[e];
    const float4* qp = (const float4*)(q  + (size_t)d * HIDDEN + h * DHEAD);
    const float4* kp = (const float4*)(kt + (size_t)s * HIDDEN + h * DHEAD);
    float acc = 0.f;
#pragma unroll
    for (int i = 0; i < 4; ++i) {
        float4 a = qp[i], b = kp[i];
        acc += a.x * b.x + a.y * b.y + a.z * b.z + a.w * b.w;
    }
    acc *= prel[h] * ATT_SCALE;
    alpha[t] = acc;
    atomicMaxFloat(maxb + (size_t)d * HEADS + h, acc);
}

// pass 2: ex = exp(alpha - max[dst,h]) ; segment sum of ex
__global__ __launch_bounds__(256)
void edge_expsum(float* __restrict__ alpha, const float* __restrict__ maxb,
                 float* __restrict__ denb, const int* __restrict__ dst, int nE) {
    int t = blockIdx.x * blockDim.x + threadIdx.x;
    if (t >= nE * HEADS) return;
    int e = t >> 3, h = t & 7;
    int d = dst[e];
    float ex = __expf(alpha[t] - maxb[(size_t)d * HEADS + h]);
    alpha[t] = ex;
    atomicAddF32(denb + (size_t)d * HEADS + h, ex);
}

// pass 3: out[dst,h,:] += (ex/den) * vt[src,h,:]
__global__ __launch_bounds__(256)
void edge_accum(const float* __restrict__ alpha, const float* __restrict__ denb,
                const float* __restrict__ vt,
                const int* __restrict__ src, const int* __restrict__ dst,
                float* __restrict__ outm, int nE) {
    int t = blockIdx.x * blockDim.x + threadIdx.x;
    if (t >= nE * HEADS) return;
    int e = t >> 3, h = t & 7;
    int s = src[e], d = dst[e];
    float attn = alpha[t] / denb[(size_t)d * HEADS + h];
    const float* vp = vt   + (size_t)s * HIDDEN + h * DHEAD;
    float*       op = outm + (size_t)d * HIDDEN + h * DHEAD;
#pragma unroll
    for (int i = 0; i < 16; ++i) atomicAddF32(op + i, attn * vp[i]);
}

// ---------------- host orchestration ----------------
extern "C" void kernel_launch(void* const* d_in, const int* in_sizes, int n_in,
                              void* d_out, int out_size, void* d_ws, size_t ws_size,
                              hipStream_t stream) {
    (void)in_sizes; (void)n_in; (void)out_size; (void)ws_size;
    const float* x_q    = (const float*)d_in[0];
    const float* x_a    = (const float*)d_in[1];
    const float* lin0_wq= (const float*)d_in[2];
    const float* lin0_bq= (const float*)d_in[3];
    const float* lin0_wa= (const float*)d_in[4];
    const float* lin0_ba= (const float*)d_in[5];
    const float* kw_q   = (const float*)d_in[6];  const float* kb_q = (const float*)d_in[7];
    const float* qw_q   = (const float*)d_in[8];  const float* qb_q = (const float*)d_in[9];
    const float* vw_q   = (const float*)d_in[10]; const float* vb_q = (const float*)d_in[11];
    const float* aw_q   = (const float*)d_in[12]; const float* ab_q = (const float*)d_in[13];
    const float* kw_a   = (const float*)d_in[14]; const float* kb_a = (const float*)d_in[15];
    const float* qw_a   = (const float*)d_in[16]; const float* qb_a = (const float*)d_in[17];
    const float* vw_a   = (const float*)d_in[18]; const float* vb_a = (const float*)d_in[19];
    const float* aw_a   = (const float*)d_in[20]; const float* ab_a = (const float*)d_in[21];
    const float* skip_q = (const float*)d_in[22];
    const float* skip_a = (const float*)d_in[23];
    const float* a_rel  = (const float*)d_in[24];
    const float* m_rel  = (const float*)d_in[25];
    const float* p_rel  = (const float*)d_in[26];
    const float* lin_w  = (const float*)d_in[27];
    const float* lin_b  = (const float*)d_in[28];
    const int*   ei_qa  = (const int*)d_in[29];
    const int*   ei_aq  = (const int*)d_in[30];
    float* out = (float*)d_out;

    char* p = (char*)d_ws;
    auto alloc = [&](size_t bytes) -> char* {
        char* r = p; p += (bytes + 255) & ~(size_t)255; return r;
    };
    const size_t NODEB = (size_t)NQN * HIDDEN * sizeof(float); // 25.6 MB (NQN == NAN_)
    float* XQ   = (float*)alloc(NODEB);
    float* XA   = (float*)alloc(NODEB);
    float* KT   = (float*)alloc(NODEB);
    float* VT   = (float*)alloc(NODEB);
    float* QD   = (float*)alloc(NODEB);
    float* OUTA = (float*)alloc(NODEB);
    float* OUTQ = (float*)alloc(NODEB);
    float* ALPHA= (float*)alloc((size_t)EEDGES * HEADS * sizeof(float));
    float* MAXB = (float*)alloc((size_t)NQN * HEADS * sizeof(float));
    float* DENB = (float*)alloc((size_t)NQN * HEADS * sizeof(float));
    __bf16* WK  = (__bf16*)alloc(128 * 128 * sizeof(__bf16));
    __bf16* WV  = (__bf16*)alloc(128 * 128 * sizeof(__bf16));
    __bf16* WQm = (__bf16*)alloc(128 * 128 * sizeof(__bf16));
    __bf16* WA  = (__bf16*)alloc(128 * 128 * sizeof(__bf16));
    __bf16* WL  = (__bf16*)alloc(128 * 64 * sizeof(__bf16));
    float* BK   = (float*)alloc(128 * sizeof(float));
    float* BV   = (float*)alloc(128 * sizeof(float));

    const int NB_CV   = (128 * 128 + 255) / 256;
    const int NB_FUSE = (129 * 128 + 255) / 256;
    const int NB_EH   = (EEDGES * HEADS + 255) / 256;
    const int NB_INIT = (NQN * HIDDEN + 255) / 256;

    auto gemm = [&](const float* A, const __bf16* W, const float* bias, float* C,
                    const float* skipOld, const float* gate, int M, int N,
                    int preGelu, int postRelu) {
        gemm_bias_act<<<dim3(M / 16), dim3(256), 0, stream>>>(A, W, bias, C, skipOld, gate,
                                                              N, preGelu, postRelu);
    };

    // input projection + relu
    conv_bf16<<<NB_CV, 256, 0, stream>>>(lin0_wq, WK, 128 * 128);
    gemm(x_q, WK, lin0_bq, XQ, nullptr, nullptr, NQN, 128, 0, 1);
    conv_bf16<<<NB_CV, 256, 0, stream>>>(lin0_wa, WK, 128 * 128);
    gemm(x_a, WK, lin0_ba, XA, nullptr, nullptr, NAN_, 128, 0, 1);

    for (int l = 0; l < 2; ++l) {
        // ---- edge type 0: question -> answer (src=q, dst=a) ----
        fuse_rel<<<NB_FUSE, 256, 0, stream>>>(kw_q + l * 16384, kb_q + l * 128,
                                              a_rel + (size_t)(l * 2 + 0) * 2048, WK, BK);
        fuse_rel<<<NB_FUSE, 256, 0, stream>>>(vw_q + l * 16384, vb_q + l * 128,
                                              m_rel + (size_t)(l * 2 + 0) * 2048, WV, BV);
        conv_bf16<<<NB_CV, 256, 0, stream>>>(qw_a + l * 16384, WQm, 128 * 128);
        gemm(XQ, WK, BK, KT, nullptr, nullptr, NQN, 128, 0, 0);
        gemm(XQ, WV, BV, VT, nullptr, nullptr, NQN, 128, 0, 0);
        gemm(XA, WQm, qb_a + l * 128, QD, nullptr, nullptr, NAN_, 128, 0, 0);
        init_msg<<<NB_INIT, 256, 0, stream>>>(OUTA, MAXB, DENB, NAN_);
        edge_alpha_max<<<NB_EH, 256, 0, stream>>>(QD, KT, ei_qa, ei_qa + EEDGES,
                                                  p_rel + (l * 2 + 0) * 8, ALPHA, MAXB, EEDGES);
        edge_expsum<<<NB_EH, 256, 0, stream>>>(ALPHA, MAXB, DENB, ei_qa + EEDGES, EEDGES);
        edge_accum<<<NB_EH, 256, 0, stream>>>(ALPHA, DENB, VT, ei_qa, ei_qa + EEDGES,
                                              OUTA, EEDGES);

        // ---- edge type 1: answer -> question (src=a, dst=q) ----
        fuse_rel<<<NB_FUSE, 256, 0, stream>>>(kw_a + l * 16384, kb_a + l * 128,
                                              a_rel + (size_t)(l * 2 + 1) * 2048, WK, BK);
        fuse_rel<<<NB_FUSE, 256, 0, stream>>>(vw_a + l * 16384, vb_a + l * 128,
                                              m_rel + (size_t)(l * 2 + 1) * 2048, WV, BV);
        conv_bf16<<<NB_CV, 256, 0, stream>>>(qw_q + l * 16384, WQm, 128 * 128);
        gemm(XA, WK, BK, KT, nullptr, nullptr, NAN_, 128, 0, 0);
        gemm(XA, WV, BV, VT, nullptr, nullptr, NAN_, 128, 0, 0);
        gemm(XQ, WQm, qb_q + l * 128, QD, nullptr, nullptr, NQN, 128, 0, 0);
        init_msg<<<NB_INIT, 256, 0, stream>>>(OUTQ, MAXB, DENB, NQN);
        edge_alpha_max<<<NB_EH, 256, 0, stream>>>(QD, KT, ei_aq, ei_aq + EEDGES,
                                                  p_rel + (l * 2 + 1) * 8, ALPHA, MAXB, EEDGES);
        edge_expsum<<<NB_EH, 256, 0, stream>>>(ALPHA, MAXB, DENB, ei_aq + EEDGES, EEDGES);
        edge_accum<<<NB_EH, 256, 0, stream>>>(ALPHA, DENB, VT, ei_aq, ei_aq + EEDGES,
                                              OUTQ, EEDGES);

        // ---- update: x = g * (gelu(out) @ aw + ab) + (1-g) * x  (in-place safe) ----
        conv_bf16<<<NB_CV, 256, 0, stream>>>(aw_q + l * 16384, WA, 128 * 128);
        gemm(OUTQ, WA, ab_q + l * 128, XQ, XQ, skip_q + l, NQN, 128, 1, 0);
        conv_bf16<<<NB_CV, 256, 0, stream>>>(aw_a + l * 16384, WA, 128 * 128);
        gemm(OUTA, WA, ab_a + l * 128, XA, XA, skip_a + l, NAN_, 128, 1, 0);
    }

    // final projection into d_out: rows [0,NQ) = question, [NQ,NQ+NA) = answer
    conv_bf16<<<(128 * 64 + 255) / 256, 256, 0, stream>>>(lin_w, WL, 128 * 64);
    gemm(XQ, WL, lin_b, out,                    nullptr, nullptr, NQN, 64, 0, 0);
    gemm(XA, WL, lin_b, out + (size_t)NQN * 64, nullptr, nullptr, NAN_, 64, 0, 0);
}
